// Decoder_62947040690578
// MI455X (gfx1250) — matrix operand order repaired and checked by reading, not compiled, not run on hardware
//
#include <hip/hip_runtime.h>
#include <stdint.h>

// ---------------------------------------------------------------------------
// Types for WMMA bf16 path (gfx1250 / CDNA5, wave32)
// ---------------------------------------------------------------------------
typedef __attribute__((ext_vector_type(16))) __bf16   v16bf;
typedef __attribute__((ext_vector_type(8)))  float    v8f;
typedef __attribute__((ext_vector_type(4)))  unsigned u32x4;
typedef __attribute__((ext_vector_type(4)))  float    fvec4;

static __device__ __forceinline__ unsigned short f32_to_bf16(float f) {
  unsigned u = __float_as_uint(f);
  unsigned r = u + 0x7FFFu + ((u >> 16) & 1u);   // round-to-nearest-even
  return (unsigned short)(r >> 16);
}

// ---------------------------------------------------------------------------
// Generic utility kernels
// ---------------------------------------------------------------------------
__global__ void k_zero(float* __restrict__ p, long long n) {
  long long i = (long long)blockIdx.x * blockDim.x + threadIdx.x;
  if (i < n) p[i] = 0.f;
}

// f32 -> bf16 with zero padding of rows [M, Mpad)
__global__ void k_cvt_bf16(const float* __restrict__ x, unsigned short* __restrict__ o,
                           int M, int Mpad, int K) {
  int idx = blockIdx.x * 256 + threadIdx.x;
  int total = Mpad * K;
  if (idx >= total) return;
  int row = idx / K;
  float v = (row < M) ? x[idx] : 0.f;
  o[idx] = f32_to_bf16(v);
}

// Pack Wcat = [W_top - W_bot | W_bot] (bf16) in WMMA-B fragment-major order:
// Bpk[((nt*KC + kc)*32 + lane)*16 + e] = Wcat[kc*32 + (e>>3)*16 + (lane>>4)*8 + (e&7)][nt*16 + (lane&15)]
__global__ void k_prep_w(const float* __restrict__ W, unsigned short* __restrict__ Bpk,
                         int Cin, int Cout) {
  int Ncols = 2 * Cout;
  int total = Cin * Ncols;
  int idx = blockIdx.x * 256 + threadIdx.x;
  if (idx >= total) return;
  int e    = idx & 15;
  int lane = (idx >> 4) & 31;
  int KC   = Cin >> 5;
  int frag = idx >> 9;                 // nt*KC + kc
  int kc = frag % KC, nt = frag / KC;
  int k = (kc << 5) + ((e >> 3) << 4) + ((lane >> 4) << 3) + (e & 7);
  int j = (nt << 4) + (lane & 15);
  float v;
  if (j < Cout) v = W[k * Cout + j] - W[(Cin + k) * Cout + j];   // W_top - W_bot
  else          v = W[(Cin + k) * Cout + (j - Cout)];            // W_bot
  Bpk[idx] = f32_to_bf16(v);
}

// ---------------------------------------------------------------------------
// WMMA GEMM: D[Mpad, Ncols] = A[Mpad, K](bf16 row-major) @ Wcat[K, Ncols]
// One 16x16 tile per wave; K-loop in steps of 32 accumulating in f32.
// ---------------------------------------------------------------------------
__global__ void k_gemm_wmma(const unsigned short* __restrict__ A,
                            const unsigned short* __restrict__ Bpk,
                            float* __restrict__ D,
                            int Mtiles, int K, int Ncols) {
  int wave = threadIdx.x >> 5;
  int lane = threadIdx.x & 31;
  int Ntiles = Ncols >> 4;
  int tile = blockIdx.x * (blockDim.x >> 5) + wave;
  if (tile >= Mtiles * Ntiles) return;          // whole wave exits: EXEC stays all-ones
  int mt = tile / Ntiles, nt = tile % Ntiles;

  int row   = (mt << 4) + (lane & 15);
  int kbase = (lane >> 4) << 3;                 // 0 or 8
  const unsigned short* arow = A + (size_t)row * K;
  int KC = K >> 5;
  const unsigned short* bbase = Bpk + ((size_t)nt * KC * 32 + lane) * 16;

  v8f acc = {0.f, 0.f, 0.f, 0.f, 0.f, 0.f, 0.f, 0.f};
  for (int kc = 0; kc < KC; ++kc) {
    union { u32x4 q[2]; v16bf v; } fa, fb;
    int k0 = kc << 5;
    fa.q[0] = *reinterpret_cast<const u32x4*>(arow + k0 + kbase);        // K = k0+kbase..+7
    fa.q[1] = *reinterpret_cast<const u32x4*>(arow + k0 + 16 + kbase);   // K = k0+16+kbase..+7
    const unsigned short* bf = bbase + (size_t)kc * (32 * 16);
    fb.q[0] = *reinterpret_cast<const u32x4*>(bf);
    fb.q[1] = *reinterpret_cast<const u32x4*>(bf + 8);
    acc = __builtin_amdgcn_wmma_f32_16x16x32_bf16(false, fa.v, false, fb.v,
                                                  (short)0, acc, false, false);
  }
  int col = (nt << 4) + (lane & 15);
  int r0  = (lane >> 4) << 3;
  float* drow = D + (size_t)((mt << 4) + r0) * Ncols + col;
#pragma unroll
  for (int r = 0; r < 8; ++r) drow[(size_t)r * Ncols] = acc[r];
}

// ---------------------------------------------------------------------------
// Per-edge: agg[dst] += relu(P[dst] + Q[src] + bias); 4 channels per thread
// PQ layout: [row, 0..C-1]=P , [row, C..2C-1]=Q
// ---------------------------------------------------------------------------
__global__ void k_edge(const float* __restrict__ PQ, const float* __restrict__ bias,
                       const int* __restrict__ ei, int E, int C, int Nn, int Bc,
                       float* __restrict__ agg) {
  int C4 = C >> 2;
  long long total = (long long)Bc * E * C4;
  long long idx = (long long)blockIdx.x * 256 + threadIdx.x;
  if (idx >= total) return;
  int c4 = (int)(idx % C4);
  long long t = idx / C4;
  int e = (int)(t % E);
  int b = (int)(t / E);
  int src = ei[e], dst = ei[E + e];
  int Ncols = C << 1;
  int c = c4 << 2;
  fvec4 p  = *reinterpret_cast<const fvec4*>(PQ + (size_t)(b * Nn + dst) * Ncols + c);
  fvec4 q  = *reinterpret_cast<const fvec4*>(PQ + (size_t)(b * Nn + src) * Ncols + C + c);
  fvec4 bb = *reinterpret_cast<const fvec4*>(bias + c);
  float* a = agg + (size_t)(b * Nn + dst) * C + c;
#pragma unroll
  for (int i = 0; i < 4; ++i) {
    float v = p[i] + q[i] + bb[i];
    v = v > 0.f ? v : 0.f;
    atomicAdd(a + i, v);
  }
}

__global__ void k_deg(const int* __restrict__ ei, int E, float* __restrict__ deg) {
  int e = blockIdx.x * 256 + threadIdx.x;
  if (e >= E) return;
  atomicAdd(deg + ei[E + e], 1.f);
}

__global__ void k_norm(const float* __restrict__ agg, const float* __restrict__ deg,
                       float* __restrict__ xo, int M, int C, int Nn) {
  int idx = blockIdx.x * 256 + threadIdx.x;
  if (idx >= M * C) return;
  int n = (idx / C) % Nn;
  xo[idx] = agg[idx] / fmaxf(deg[n], 1.f);
}

// Unpool scatter: dst[b*upN + idx[i]] = src[b*n + i] (dst pre-zeroed)
__global__ void k_scatter(const float* __restrict__ src, const int* __restrict__ idxarr,
                          float* __restrict__ dst, int n, int upN, int C, int Bc) {
  long long total = (long long)Bc * n * C;
  long long idx = (long long)blockIdx.x * 256 + threadIdx.x;
  if (idx >= total) return;
  int c = (int)(idx % C);
  long long t = idx / C;
  int i = (int)(t % n);
  int b = (int)(t / n);
  dst[((size_t)b * upN + idxarr[i]) * C + c] = src[((size_t)b * n + i) * C + c];
}

__global__ void k_resid(const float* __restrict__ y, const float* __restrict__ s,
                        float* __restrict__ o, int total) {
  int i = blockIdx.x * 256 + threadIdx.x;
  if (i >= total) return;
  float v = y[i] + s[i];
  o[i] = v > 0.f ? v : 0.01f * v;
}

// ---------------------------------------------------------------------------
// Latent up-MLP: t[b,l,k] = lrelu(z[b,l]*W1[k]+b1[k]); x0[b,n,l] = t[b,l,:]@W2[:,n]+b2[n]
// ---------------------------------------------------------------------------
__global__ void k_latA(const float* __restrict__ z, const float* __restrict__ W1,
                       const float* __restrict__ b1, float* __restrict__ t) {
  int idx = blockIdx.x * 256 + threadIdx.x;   // B*128*64 = 16384
  if (idx >= 2 * 128 * 64) return;
  int k = idx & 63;
  int l = (idx >> 6) & 127;
  int b = idx >> 13;
  float v = z[b * 128 + l] * W1[k] + b1[k];
  t[idx] = v > 0.f ? v : 0.01f * v;
}

__global__ void k_latB(const float* __restrict__ t, const float* __restrict__ W2,
                       const float* __restrict__ b2, float* __restrict__ x0) {
  const int N2 = 6250;
  int idx = blockIdx.x * 256 + threadIdx.x;   // B*N2*16 = 200000
  if (idx >= 2 * N2 * 16) return;
  int lg = idx & 15;
  int n = (idx >> 4) % N2;
  int b = idx / (N2 * 16);
  int l0 = lg << 3;
  float acc[8] = {};
  const float* tb = t + (size_t)(b * 128 + l0) * 64;
  for (int k = 0; k < 64; ++k) {
    float w = W2[(size_t)k * N2 + n];
#pragma unroll
    for (int j = 0; j < 8; ++j) acc[j] += tb[j * 64 + k] * w;
  }
  float bb = b2[n];
  float* xo = x0 + ((size_t)(b * N2 + n)) * 128 + l0;
#pragma unroll
  for (int j = 0; j < 8; ++j) xo[j] = acc[j] + bb;
}

// ---------------------------------------------------------------------------
// Decoder MLP + LayerNorm(3): weights staged in LDS
// ---------------------------------------------------------------------------
__global__ void k_decode(const float* __restrict__ X, const float* __restrict__ Wd1,
                         const float* __restrict__ bd1, const float* __restrict__ Wd2,
                         const float* __restrict__ bd2, const float* __restrict__ gamma,
                         const float* __restrict__ beta, float* __restrict__ out, int M) {
  __shared__ float sW1[64 * 32];
  __shared__ float sW2[32 * 3];
  __shared__ float sb1[32], sb2[3], sg[3], sbt[3];
  for (int i = threadIdx.x; i < 64 * 32; i += blockDim.x) sW1[i] = Wd1[i];
  for (int i = threadIdx.x; i < 32 * 3; i += blockDim.x) sW2[i] = Wd2[i];
  if (threadIdx.x < 32) sb1[threadIdx.x] = bd1[threadIdx.x];
  if (threadIdx.x < 3) {
    sb2[threadIdx.x] = bd2[threadIdx.x];
    sg[threadIdx.x] = gamma[threadIdx.x];
    sbt[threadIdx.x] = beta[threadIdx.x];
  }
  __syncthreads();
  int row = blockIdx.x * blockDim.x + threadIdx.x;
  if (row >= M) return;
  float x[64];
  const float* xr = X + (size_t)row * 64;
#pragma unroll
  for (int k = 0; k < 64; ++k) x[k] = xr[k];
  float o0 = sb2[0], o1 = sb2[1], o2 = sb2[2];
  for (int j = 0; j < 32; ++j) {
    float s = sb1[j];
#pragma unroll
    for (int k = 0; k < 64; ++k) s += x[k] * sW1[k * 32 + j];
    s = s > 0.f ? s : 0.01f * s;      // leaky relu
    o0 += s * sW2[j * 3 + 0];
    o1 += s * sW2[j * 3 + 1];
    o2 += s * sW2[j * 3 + 2];
  }
  float mu = (o0 + o1 + o2) * (1.f / 3.f);
  float d0 = o0 - mu, d1 = o1 - mu, d2 = o2 - mu;
  float var = (d0 * d0 + d1 * d1 + d2 * d2) * (1.f / 3.f);
  float inv = rsqrtf(var + 1e-5f);
  float* op = out + (size_t)row * 3;
  op[0] = d0 * inv * sg[0] + sbt[0];
  op[1] = d1 * inv * sg[1] + sbt[1];
  op[2] = d2 * inv * sg[2] + sbt[2];
}

// ---------------------------------------------------------------------------
// Host orchestration
// ---------------------------------------------------------------------------
extern "C" void kernel_launch(void* const* d_in, const int* in_sizes, int n_in,
                              void* d_out, int out_size, void* d_ws, size_t ws_size,
                              hipStream_t stream) {
  (void)in_sizes; (void)n_in; (void)out_size; (void)ws_size;
  constexpr int B = 2, N0 = 100000, N1 = 25000, N2 = 6250;
  constexpr int E0 = 800000, E1 = 200000, E2 = 50000;

  const float* z      = (const float*)d_in[0];
  const int*   g0     = (const int*)d_in[1];
  const int*   g1     = (const int*)d_in[2];
  const int*   g2     = (const int*)d_in[3];
  const int*   m_id0  = (const int*)d_in[4];
  const int*   m_id1  = (const int*)d_in[5];
  const float* W_up1  = (const float*)d_in[6];
  const float* b_up1  = (const float*)d_in[7];
  const float* W_up2  = (const float*)d_in[8];
  const float* b_up2  = (const float*)d_in[9];
  const float* Wb     = (const float*)d_in[10];
  const float* bb     = (const float*)d_in[11];
  const float* l0_W1  = (const float*)d_in[12];
  const float* l0_b1  = (const float*)d_in[13];
  const float* l0_W2  = (const float*)d_in[14];
  const float* l0_b2  = (const float*)d_in[15];
  const float* l0_Wsk = (const float*)d_in[16];
  const float* l0_bsk = (const float*)d_in[17];
  const float* l1_W1  = (const float*)d_in[18];
  const float* l1_b1  = (const float*)d_in[19];
  const float* l1_W2  = (const float*)d_in[20];
  const float* l1_b2  = (const float*)d_in[21];
  const float* l1_Wsk = (const float*)d_in[22];
  const float* l1_bsk = (const float*)d_in[23];
  const float* Wf     = (const float*)d_in[24];
  const float* bfv    = (const float*)d_in[25];
  const float* Wd1    = (const float*)d_in[26];
  const float* bd1    = (const float*)d_in[27];
  const float* Wd2    = (const float*)d_in[28];
  const float* bd2    = (const float*)d_in[29];
  const float* gamma  = (const float*)d_in[30];
  const float* beta   = (const float*)d_in[31];

  // ---- bump allocator over d_ws (~470 MB total) ----
  size_t off = 0;
  auto alloc = [&](size_t bytes) -> char* {
    size_t a = (off + 255) & ~(size_t)255;
    off = a + bytes;
    return (char*)d_ws + a;
  };
  float* t_lat = (float*)alloc((size_t)2 * 128 * 64 * 4);
  float* x0    = (float*)alloc((size_t)B * N2 * 128 * 4);
  float* xc1   = (float*)alloc((size_t)B * N2 * 256 * 4);
  float* U     = (float*)alloc((size_t)B * N0 * 128 * 4);   // unpool buf / final conv out
  float* T     = (float*)alloc((size_t)B * N1 * 64 * 4);
  float* X1    = (float*)alloc((size_t)B * N1 * 128 * 4);
  float* X0    = (float*)alloc((size_t)B * N0 * 64 * 4);
  float* S     = (float*)alloc((size_t)B * N0 * 64 * 4);    // skip storage (max)
  float* PQ    = (float*)alloc((size_t)B * N0 * 128 * 4);
  float* AGG   = (float*)alloc((size_t)B * N0 * 64 * 4);
  unsigned short* XBF = (unsigned short*)alloc((size_t)B * N0 * 128 * 2);
  unsigned short* WPK = (unsigned short*)alloc((size_t)131072 * 2);
  float* deg0 = (float*)alloc((size_t)N0 * 4);
  float* deg1 = (float*)alloc((size_t)N1 * 4);
  float* deg2 = (float*)alloc((size_t)N2 * 4);

  auto zero = [&](float* p, long long n) {
    k_zero<<<(unsigned)((n + 255) / 256), 256, 0, stream>>>(p, n);
  };

  // EdgeConv: xout[M,Cout] = mean-agg over edges of relu([xi, xj-xi]@W + b)
  auto run_conv = [&](const float* xin, float* xout, const float* W, const float* bias,
                      int Cin, int Cout, const int* ei, int E, int Nn, const float* deg) {
    int M = B * Nn;
    int Mpad = (M + 15) & ~15;
    int Ncols = 2 * Cout;
    int totW = Cin * Ncols;
    k_prep_w<<<(totW + 255) / 256, 256, 0, stream>>>(W, WPK, Cin, Cout);
    int totC = Mpad * Cin;
    k_cvt_bf16<<<(totC + 255) / 256, 256, 0, stream>>>(xin, XBF, M, Mpad, Cin);
    int tiles = (Mpad / 16) * (Ncols / 16);
    k_gemm_wmma<<<(tiles + 3) / 4, 128, 0, stream>>>(XBF, WPK, PQ, Mpad / 16, Cin, Ncols);
    zero(AGG, (long long)M * Cout);
    long long totE = (long long)B * E * (Cout / 4);
    k_edge<<<(unsigned)((totE + 255) / 256), 256, 0, stream>>>(PQ, bias, ei, E, Cout, Nn, B, AGG);
    int totN = M * Cout;
    k_norm<<<(totN + 255) / 256, 256, 0, stream>>>(AGG, deg, xout, M, Cout, Nn);
  };

  auto unpool = [&](const float* src, const int* idxarr, float* dst, int n, int upN, int C) {
    zero(dst, (long long)B * upN * C);
    long long tot = (long long)B * n * C;
    k_scatter<<<(unsigned)((tot + 255) / 256), 256, 0, stream>>>(src, idxarr, dst, n, upN, C, B);
  };

  // ---- degrees (shared across batch & convs per graph) ----
  zero(deg0, N0); zero(deg1, N1); zero(deg2, N2);
  k_deg<<<(E0 + 255) / 256, 256, 0, stream>>>(g0, E0, deg0);
  k_deg<<<(E1 + 255) / 256, 256, 0, stream>>>(g1, E1, deg1);
  k_deg<<<(E2 + 255) / 256, 256, 0, stream>>>(g2, E2, deg2);

  // ---- latent -> x0 [B,N2,128] ----
  k_latA<<<64, 256, 0, stream>>>(z, W_up1, b_up1, t_lat);
  k_latB<<<(2 * N2 * 16 + 255) / 256, 256, 0, stream>>>(t_lat, W_up2, b_up2, x0);

  // ---- mpl_bottom: conv(x0, g2) -> xc1 [B,N2,256] ----
  run_conv(x0, xc1, Wb, bb, 128, 256, g2, E2, N2, deg2);

  // ---- Res_up layer 0 (N2 -> N1) ----
  unpool(xc1, m_id1, U, N2, N1, 256);
  run_conv(U, S, l0_Wsk, l0_bsk, 256, 128, g1, E1, N1, deg1);   // skip [B*N1,128]
  run_conv(xc1, T, l0_W1, l0_b1, 256, 64, g2, E2, N2, deg2);    // y1 [B*N2,64]
  unpool(T, m_id1, U, N2, N1, 64);
  run_conv(U, X1, l0_W2, l0_b2, 64, 128, g1, E1, N1, deg1);     // y2 [B*N1,128]
  k_resid<<<(B * N1 * 128 + 255) / 256, 256, 0, stream>>>(X1, S, X1, B * N1 * 128);

  // ---- Res_up layer 1 (N1 -> N0) ----
  unpool(X1, m_id0, U, N1, N0, 128);
  run_conv(U, S, l1_Wsk, l1_bsk, 128, 64, g0, E0, N0, deg0);    // skip [B*N0,64]
  run_conv(X1, T, l1_W1, l1_b1, 128, 64, g1, E1, N1, deg1);     // y1 [B*N1,64]
  unpool(T, m_id0, U, N1, N0, 64);
  run_conv(U, X0, l1_W2, l1_b2, 64, 64, g0, E0, N0, deg0);      // y2 [B*N0,64]
  k_resid<<<(B * N0 * 64 + 255) / 256, 256, 0, stream>>>(X0, S, X0, B * N0 * 64);

  // ---- final edge conv (reuse U as output; dead otherwise) ----
  run_conv(X0, U, Wf, bfv, 64, 64, g0, E0, N0, deg0);

  // ---- decoder MLP + LayerNorm -> d_out [B*N0, 3] ----
  k_decode<<<(B * N0 + 127) / 128, 128, 0, stream>>>(U, Wd1, bd1, Wd2, bd2, gamma, beta,
                                                     (float*)d_out, B * N0);
}
